// MemoryTransformerDecoderLayer_22917945492167
// MI455X (gfx1250) — matrix operand order repaired and checked by reading, not compiled, not run on hardware
//
#include <hip/hip_runtime.h>
#include <hip/hip_bf16.h>
#include <math.h>
#include <stdint.h>

// ---------------------------------------------------------------------------
// MemoryTransformerDecoderLayer for MI455X (gfx1250, wave32, WMMA + async LDS)
// B=4, T=1024, S=2048, D=512, H=8, dh=64, DFF=2048, fp32 I/O, f16 matmul paths
// ---------------------------------------------------------------------------

typedef _Float16 f16;
typedef __attribute__((ext_vector_type(16))) _Float16 v16h;
typedef __attribute__((ext_vector_type(8)))  float    v8f;

#define DM   512          // model dim
#define HN   8            // heads
#define DHD  64           // head dim
#define EPSL 1e-5f

// ---------------- CDNA5 async copy helpers ---------------------------------

// 32B global -> LDS async copy (2 x b128), tracked by ASYNCcnt.
// INST_OFFSET applies to both LDS and global addresses (ISA 08 §4.4).
static __device__ __forceinline__ void async_cp32(f16* lds_dst,
                                                  const f16* gsrc) {
  const unsigned l = (unsigned)(uintptr_t)lds_dst;        // LDS byte offset
  const unsigned long long g = (unsigned long long)(uintptr_t)gsrc;
  asm volatile(
      "global_load_async_to_lds_b128 %0, %1, off\n\t"
      "global_load_async_to_lds_b128 %0, %1, off offset:16"
      :: "v"(l), "v"(g) : "memory");
}

static __device__ __forceinline__ void wait_async0() {
  asm volatile("s_wait_asynccnt 0" ::: "memory");
}

// ---------------- WMMA helpers (CDNA5 layouts, ISA 7.12.2) -----------------

static __device__ __forceinline__ v16h load_frag16(const f16* p0, const f16* p1) {
  union { v16h v; uint4 u[2]; } r;
  r.u[0] = *(const uint4*)p0;   // 8 halves
  r.u[1] = *(const uint4*)p1;   // 8 halves
  return r.v;
}

static __device__ __forceinline__ v8f wmma16(v16h a, v16h b, v8f c) {
  // D = A(16x32 f16) * B(32x16 f16) + C(16x16 f32)
  return __builtin_amdgcn_wmma_f32_16x16x32_f16(false, a, false, b, (short)0, c,
                                                false, false);
}

static __device__ __forceinline__ v8f zero8() {
  v8f z;
#pragma unroll
  for (int i = 0; i < 8; ++i) z[i] = 0.0f;
  return z;
}

// A fragment: 16x32 row-major tile, leading dim `ld` (halves).
// lane L: row = L&15; VGPR0-3 = K[hi*8 .. +7], VGPR4-7 = K[hi*8+16 .. +23]
static __device__ __forceinline__ v16h frag_a(const f16* tile, int ld, int kbase,
                                              int lane) {
  const int row = lane & 15, hi = lane >> 4;
  const f16* p0 = tile + row * ld + kbase + hi * 8;
  return load_frag16(p0, p0 + 16);
}

// B fragment: Bt stored [n][k] row-major, leading dim `ld` (halves).
// lane L: col n = L&15; VGPR0-3 = K[hi*8 .. +7], VGPR4-7 = K[hi*8+16 .. +23]
static __device__ __forceinline__ v16h frag_b(const f16* tileT, int ld, int kbase,
                                              int lane) {
  const int n = lane & 15, hi = lane >> 4;
  const f16* p0 = tileT + n * ld + kbase + hi * 8;
  return load_frag16(p0, p0 + 16);
}

// ---------------- elementwise convert f32 -> f16 ---------------------------

__global__ void cvt_f32_f16(const float* __restrict__ in, f16* __restrict__ out,
                            size_t n) {
  for (size_t i = (size_t)blockIdx.x * blockDim.x + threadIdx.x; i < n;
       i += (size_t)gridDim.x * blockDim.x)
    out[i] = (f16)in[i];
}

// ---------------- bias preparation -----------------------------------------

__global__ __launch_bounds__(256) void bias_prep_kernel(
    const float* __restrict__ ts, const unsigned char* __restrict__ tkpm,
    float* __restrict__ qslog, float* __restrict__ quo, int T) {
  const int b = blockIdx.x, tid = threadIdx.x;
  const int PER = T / 256;  // 4
  __shared__ float smin[256], smax[256];
  __shared__ int   sany[256];
  float qsv[8]; int mkv[8];
  float lmin = 3.4e38f, lmax = -3.4e38f; int lany = 0;
#pragma unroll
  for (int i = 0; i < PER; ++i) {
    const int t = tid + i * 256;
    const int mk = tkpm[b * T + t];
    float qs = mk ? 1.0f : fmaxf(ts[b * T + t], 1e-6f);
    qsv[i] = qs; mkv[i] = mk;
    if (!mk) { lmin = fminf(lmin, qs); lmax = fmaxf(lmax, qs); lany = 1; }
  }
  smin[tid] = lmin; smax[tid] = lmax; sany[tid] = lany;
  __syncthreads();
  for (int st = 128; st > 0; st >>= 1) {
    if (tid < st) {
      smin[tid] = fminf(smin[tid], smin[tid + st]);
      smax[tid] = fmaxf(smax[tid], smax[tid + st]);
      sany[tid] |= sany[tid + st];
    }
    __syncthreads();
  }
  const float qmin = sany[0] ? smin[0] : 1.0f;
  const float qmax = sany[0] ? smax[0] : 1.0f;
  const float qr   = qmax - qmin;
#pragma unroll
  for (int i = 0; i < PER; ++i) {
    const int t = tid + i * 256;
    float u;
    if (qr < 1e-6f) u = 1.0f - fminf(fmaxf(qsv[i], 0.0f), 1.0f);
    else            u = 1.0f - (qsv[i] - qmin) / fmaxf(qr, 1e-6f);
    if (mkv[i]) u = 0.0f;
    quo[b * T + t]   = u;
    qslog[b * T + t] = __logf(qsv[i]);
  }
}

__global__ void ks_prep_kernel(const float* __restrict__ ms,
                               const unsigned char* __restrict__ mkpm,
                               float* __restrict__ ksb, int n) {
  const int i = blockIdx.x * blockDim.x + threadIdx.x;
  if (i < n) ksb[i] = mkpm[i] ? 1.0f : fmaxf(ms[i], 1e-6f);
}

// ---------------- generic WMMA GEMM: C = A * W^T + bias --------------------
// A: (M,K) f16 row-major.  W: (N,K) f16 row-major (== Bt[n][k]).
// MODE 0: f16 out.  MODE 1: f16 out + relu.  MODE 2: f32 out + residual.
// Double-buffered LDS ping-pong fed by global_load_async_to_lds_b128.

#define BM 128
#define BN 128
#define BK 32
#define LDT 40   // padded tile stride (halves): bank-conflict-free

template <int MODE>
__global__ __launch_bounds__(256) void gemm_wmma_kernel(
    const f16* __restrict__ A, const f16* __restrict__ W,
    const float* __restrict__ bias, const float* __restrict__ res,
    float* __restrict__ outF, f16* __restrict__ outH,
    int M, int N, int K) {
  __shared__ f16 As[2][BM * LDT];
  __shared__ f16 Bs[2][BN * LDT];

  const int tid  = threadIdx.x;
  const int lane = tid & 31, wave = tid >> 5;
  const int wm = wave >> 1, wn = wave & 1;       // 4 x 2 wave grid
  const int bm = blockIdx.y * BM, bn = blockIdx.x * BN;

  v8f acc[2][4];
#pragma unroll
  for (int i = 0; i < 2; ++i)
#pragma unroll
    for (int j = 0; j < 4; ++j) acc[i][j] = zero8();

  const int rA  = tid >> 1;            // 0..127 tile row for loading
  const int seg = (tid & 1) * 16;      // 0 or 16 halves

  const f16* gA = A + (size_t)(bm + rA) * K + seg;
  const f16* gB = W + (size_t)(bn + rA) * K + seg;
  f16* lA = &As[0][0] + rA * LDT + seg;
  f16* lB = &Bs[0][0] + rA * LDT + seg;
  const int bufStride = BM * LDT;

  const int nk = K / BK;
  // prologue: stage tile 0 into buffer 0
  async_cp32(lA, gA);
  async_cp32(lB, gB);

  for (int i = 0; i < nk; ++i) {
    const int cur = i & 1;
    wait_async0();        // own async copies into `cur` landed
    __syncthreads();      // everyone's copies landed; prev reads of cur^1 done
    if (i + 1 < nk) {     // stage next tile into cur^1
      async_cp32(lA + (cur ^ 1) * bufStride, gA + (i + 1) * BK);
      async_cp32(lB + (cur ^ 1) * bufStride, gB + (i + 1) * BK);
      if (i + 2 < nk) {   // keep L2 warm two tiles ahead
        __builtin_prefetch(gA + (i + 2) * BK, 0, 3);
        __builtin_prefetch(gB + (i + 2) * BK, 0, 3);
      }
    }

    const f16* Ac = &As[cur][0];
    const f16* Bc = &Bs[cur][0];
    v16h af[2], bf[4];
#pragma unroll
    for (int a = 0; a < 2; ++a)
      af[a] = frag_a(Ac + (wm * 32 + a * 16) * LDT, LDT, 0, lane);
#pragma unroll
    for (int j = 0; j < 4; ++j)
      bf[j] = frag_b(Bc + (wn * 64 + j * 16) * LDT, LDT, 0, lane);
#pragma unroll
    for (int a = 0; a < 2; ++a)
#pragma unroll
      for (int j = 0; j < 4; ++j) acc[a][j] = wmma16(af[a], bf[j], acc[a][j]);
  }

  const int hi = lane >> 4, ln = lane & 15;
#pragma unroll
  for (int i = 0; i < 2; ++i) {
#pragma unroll
    for (int j = 0; j < 4; ++j) {
      const int col = bn + wn * 64 + j * 16 + ln;
      const float bv = bias[col];
#pragma unroll
      for (int r = 0; r < 8; ++r) {
        const int row = bm + wm * 32 + i * 16 + r + 8 * hi;
        float v = acc[i][j][r] + bv;
        const size_t idx = (size_t)row * N + col;
        if (MODE == 2) {
          outF[idx] = v + res[idx];
        } else {
          if (MODE == 1) v = fmaxf(v, 0.0f);
          outH[idx] = (f16)v;
        }
      }
    }
  }
}

// ---------------- fused flash attention (per (b,h), 128 q-rows/block) ------

__global__ __launch_bounds__(256) void attn_kernel(
    const f16* __restrict__ Q, const f16* __restrict__ Kh,
    const f16* __restrict__ Vh, f16* __restrict__ Out,
    const float* __restrict__ qslog, const float* __restrict__ qu,
    const float* __restrict__ ksb, const unsigned char* __restrict__ mask,
    int Tq, int Sk, int use_bias, float scale) {
  __shared__ f16 Ksm[64 * 72];          // K tile [s][d]
  __shared__ f16 Vtm[64 * 72];          // V tile transposed [d][s]
  __shared__ f16 Psm[8 * 16 * 72];      // per-wave P tiles [m][s]

  const int tid = threadIdx.x, lane = tid & 31, wave = tid >> 5;
  const int hi = lane >> 4, ln = lane & 15;
  const int b = blockIdx.y / HN, h = blockIdx.y % HN;
  const int row0 = blockIdx.x * 128 + wave * 16;   // this wave's 16 q rows
  const int dcol = h * DHD;

  const f16* Qb = Q  + ((size_t)b * Tq) * DM + dcol;
  const f16* Kb = Kh + ((size_t)b * Sk) * DM + dcol;
  const f16* Vb = Vh + ((size_t)b * Sk) * DM + dcol;

  // Q fragments straight from global (row-major == A-frag layout)
  v16h aq[2];
#pragma unroll
  for (int kk = 0; kk < 2; ++kk) {
    const f16* p0 = Qb + (size_t)(row0 + ln) * DM + kk * 32 + hi * 8;
    aq[kk] = load_frag16(p0, p0 + 16);
  }

  float m_r[8], l_r[8], rqlog[8], rqu[8];
#pragma unroll
  for (int r = 0; r < 8; ++r) { m_r[r] = -1e30f; l_r[r] = 0.0f; }
  if (use_bias) {
#pragma unroll
    for (int r = 0; r < 8; ++r) {
      const int t = row0 + r + 8 * hi;
      rqlog[r] = qslog[b * Tq + t];
      rqu[r]   = qu[b * Tq + t];
    }
  }
  v8f accO[4];
#pragma unroll
  for (int d = 0; d < 4; ++d) accO[d] = zero8();

  f16* Pw = Psm + wave * 16 * 72;
  const int j = tid >> 2, seg = (tid & 3) * 16;

  for (int s0 = 0; s0 < Sk; s0 += 64) {
    __syncthreads();   // all waves done reading previous K/V tiles
    // stage K tile via CDNA5 async LDS copy
    async_cp32(Ksm + j * 72 + seg, Kb + (size_t)(s0 + j) * DM + seg);
    // stage V^T tile (transpose by hand)
    {
      f16 tv[16];
      const uint4* gv = (const uint4*)(Vb + (size_t)(s0 + j) * DM + seg);
      *(uint4*)(tv)     = gv[0];
      *(uint4*)(tv + 8) = gv[1];
#pragma unroll
      for (int e = 0; e < 16; ++e) Vtm[(seg + e) * 72 + j] = tv[e];
    }
    wait_async0();
    __syncthreads();

    // logits: S = Q K^T
    v8f accS[4];
#pragma unroll
    for (int nt = 0; nt < 4; ++nt) accS[nt] = zero8();
#pragma unroll
    for (int kk = 0; kk < 2; ++kk)
#pragma unroll
      for (int nt = 0; nt < 4; ++nt) {
        v16h bk = frag_b(Ksm + nt * 16 * 72, 72, kk * 32, lane);
        accS[nt] = wmma16(aq[kk], bk, accS[nt]);
      }

    // scale + analytic bias + key-padding mask; tile row-max
    float pv[4][8], tm[8];
#pragma unroll
    for (int r = 0; r < 8; ++r) tm[r] = -1e30f;
#pragma unroll
    for (int nt = 0; nt < 4; ++nt) {
      const int sidx = s0 + nt * 16 + ln;
      float kv = 1.0f; int mk = 0;
      if (use_bias) kv = ksb[b * Sk + sidx];
      if (mask)     mk = mask[b * Sk + sidx];
#pragma unroll
      for (int r = 0; r < 8; ++r) {
        float v = accS[nt][r] * scale;
        if (use_bias)
          v += rqlog[r] + __logf(fmaxf(fmaf(rqu[r], kv - 1.0f, 1.0f), 1e-6f));
        if (mk) v = -1e30f;
        pv[nt][r] = v;
        tm[r] = fmaxf(tm[r], v);
      }
    }
#pragma unroll
    for (int off = 1; off < 16; off <<= 1)
#pragma unroll
      for (int r = 0; r < 8; ++r) tm[r] = fmaxf(tm[r], __shfl_xor(tm[r], off));

    float alpha[8], rs[8];
#pragma unroll
    for (int r = 0; r < 8; ++r) {
      const float mn = fmaxf(m_r[r], tm[r]);
      alpha[r] = __expf(m_r[r] - mn);
      m_r[r] = mn;
      rs[r] = 0.0f;
    }
#pragma unroll
    for (int nt = 0; nt < 4; ++nt)
#pragma unroll
      for (int r = 0; r < 8; ++r) {
        const float p = __expf(pv[nt][r] - m_r[r]);
        rs[r] += p;
        Pw[(r + 8 * hi) * 72 + nt * 16 + ln] = (f16)p;
      }
#pragma unroll
    for (int off = 1; off < 16; off <<= 1)
#pragma unroll
      for (int r = 0; r < 8; ++r) rs[r] += __shfl_xor(rs[r], off);
#pragma unroll
    for (int r = 0; r < 8; ++r) l_r[r] = l_r[r] * alpha[r] + rs[r];

#pragma unroll
    for (int d = 0; d < 4; ++d)
#pragma unroll
      for (int r = 0; r < 8; ++r) accO[d][r] *= alpha[r];

    // O += P * V
#pragma unroll
    for (int kk = 0; kk < 2; ++kk) {
      v16h ap = frag_a(Pw, 72, kk * 32, lane);
#pragma unroll
      for (int d = 0; d < 4; ++d) {
        v16h bv = frag_b(Vtm + d * 16 * 72, 72, kk * 32, lane);
        accO[d] = wmma16(ap, bv, accO[d]);
      }
    }
  }

  // normalize + store f16 head slice
#pragma unroll
  for (int d = 0; d < 4; ++d)
#pragma unroll
    for (int r = 0; r < 8; ++r) {
      const int t = row0 + r + 8 * hi;
      const int col = dcol + d * 16 + ln;
      Out[((size_t)b * Tq + t) * DM + col] = (f16)(accO[d][r] / l_r[r]);
    }
}

// ---------------- LayerNorm (wave per row, D=512) --------------------------

__global__ __launch_bounds__(256) void ln_kernel(
    const float* __restrict__ y, const float* __restrict__ g,
    const float* __restrict__ bb, float* __restrict__ xo,
    f16* __restrict__ xh, int rows) {
  const int wave = threadIdx.x >> 5, lane = threadIdx.x & 31;
  const int row = blockIdx.x * 8 + wave;
  if (row >= rows) return;
  const float* p = y + (size_t)row * DM;
  float v[16], s = 0.0f, ss = 0.0f;
#pragma unroll
  for (int i = 0; i < 16; ++i) {
    const float t = p[lane + i * 32];
    v[i] = t; s += t; ss += t * t;
  }
#pragma unroll
  for (int off = 16; off > 0; off >>= 1) {
    s  += __shfl_xor(s, off);
    ss += __shfl_xor(ss, off);
  }
  const float mean = s * (1.0f / DM);
  const float var  = ss * (1.0f / DM) - mean * mean;
  const float inv  = rsqrtf(var + EPSL);
#pragma unroll
  for (int i = 0; i < 16; ++i) {
    const int c = lane + i * 32;
    const float o = (v[i] - mean) * inv * g[c] + bb[c];
    xo[(size_t)row * DM + c] = o;
    if (xh) xh[(size_t)row * DM + c] = (f16)o;
  }
}

// ---------------------------------------------------------------------------

extern "C" void kernel_launch(void* const* d_in, const int* in_sizes, int n_in,
                              void* d_out, int out_size, void* d_ws,
                              size_t ws_size, hipStream_t stream) {
  const int B = 4, T = 1024, S = 2048, D = DM, F = 2048;
  const int BT = B * T, BS = B * S;

  const float* tgt  = (const float*)d_in[0];
  const float* mem  = (const float*)d_in[1];
  const float* tsc  = (const float*)d_in[2];
  const float* msc  = (const float*)d_in[3];
  const unsigned char* tkpm = (const unsigned char*)d_in[4];
  const unsigned char* mkpm = (const unsigned char*)d_in[5];
  const float* sa_wq = (const float*)d_in[6];  const float* sa_bq = (const float*)d_in[7];
  const float* sa_wk = (const float*)d_in[8];  const float* sa_bk = (const float*)d_in[9];
  const float* sa_wv = (const float*)d_in[10]; const float* sa_bv = (const float*)d_in[11];
  const float* sa_wo = (const float*)d_in[12]; const float* sa_bo = (const float*)d_in[13];
  const float* ca_wq = (const float*)d_in[14]; const float* ca_bq = (const float*)d_in[15];
  const float* ca_wk = (const float*)d_in[16]; const float* ca_bk = (const float*)d_in[17];
  const float* ca_wv = (const float*)d_in[18]; const float* ca_bv = (const float*)d_in[19];
  const float* ca_wo = (const float*)d_in[20]; const float* ca_bo = (const float*)d_in[21];
  const float* w1 = (const float*)d_in[22]; const float* b1 = (const float*)d_in[23];
  const float* w2 = (const float*)d_in[24]; const float* b2 = (const float*)d_in[25];
  const float* ln1g = (const float*)d_in[26]; const float* ln1b = (const float*)d_in[27];
  const float* ln2g = (const float*)d_in[28]; const float* ln2b = (const float*)d_in[29];
  const float* ln3g = (const float*)d_in[30]; const float* ln3b = (const float*)d_in[31];

  // workspace carve-out
  char* wp = (char*)d_ws;
  auto alloc = [&](size_t bytes) -> void* {
    void* p = (void*)wp;
    wp += (bytes + 255) & ~(size_t)255;
    return p;
  };
  f16* tgt_h = (f16*)alloc((size_t)BT * D * 2);
  f16* mem_h = (f16*)alloc((size_t)BS * D * 2);
  f16* wh[10];
  const float* wsrc[10] = {sa_wq, sa_wk, sa_wv, sa_wo, ca_wq, ca_wk, ca_wv, ca_wo, w1, w2};
  size_t wn[10] = {(size_t)D*D,(size_t)D*D,(size_t)D*D,(size_t)D*D,
                   (size_t)D*D,(size_t)D*D,(size_t)D*D,(size_t)D*D,
                   (size_t)F*D,(size_t)D*F};
  for (int i = 0; i < 10; ++i) wh[i] = (f16*)alloc(wn[i] * 2);
  f16* q1  = (f16*)alloc((size_t)BT * D * 2);
  f16* k1  = (f16*)alloc((size_t)BT * D * 2);
  f16* v1  = (f16*)alloc((size_t)BT * D * 2);
  f16* at1 = (f16*)alloc((size_t)BT * D * 2);
  f16* q2  = (f16*)alloc((size_t)BT * D * 2);
  f16* k2  = (f16*)alloc((size_t)BS * D * 2);
  f16* v2  = (f16*)alloc((size_t)BS * D * 2);
  f16* at2 = (f16*)alloc((size_t)BT * D * 2);
  f16* x1h = (f16*)alloc((size_t)BT * D * 2);
  f16* x2h = (f16*)alloc((size_t)BT * D * 2);
  f16* h1h = (f16*)alloc((size_t)BT * F * 2);
  float* ytmp = (float*)alloc((size_t)BT * D * 4);
  float* x1   = (float*)alloc((size_t)BT * D * 4);
  float* x2   = (float*)alloc((size_t)BT * D * 4);
  float* qslog = (float*)alloc((size_t)BT * 4);
  float* quv   = (float*)alloc((size_t)BT * 4);
  float* ksb   = (float*)alloc((size_t)BS * 4);

  const float scale = 0.125f;  // 1/sqrt(64)

  // --- converts ---
  cvt_f32_f16<<<2048, 256, 0, stream>>>(tgt, tgt_h, (size_t)BT * D);
  cvt_f32_f16<<<2048, 256, 0, stream>>>(mem, mem_h, (size_t)BS * D);
  for (int i = 0; i < 10; ++i)
    cvt_f32_f16<<<2048, 256, 0, stream>>>(wsrc[i], wh[i], wn[i]);

  // --- bias prep ---
  bias_prep_kernel<<<B, 256, 0, stream>>>(tsc, tkpm, qslog, quv, T);
  ks_prep_kernel<<<BS / 256, 256, 0, stream>>>(msc, mkpm, ksb, BS);

  auto gemmH = [&](const f16* A, const f16* W, const float* bias, f16* oH,
                   int M, int N, int K, int relu) {
    dim3 g(N / BN, M / BM);
    if (relu)
      gemm_wmma_kernel<1><<<g, 256, 0, stream>>>(A, W, bias, nullptr, nullptr,
                                                 oH, M, N, K);
    else
      gemm_wmma_kernel<0><<<g, 256, 0, stream>>>(A, W, bias, nullptr, nullptr,
                                                 oH, M, N, K);
  };
  auto gemmF = [&](const f16* A, const f16* W, const float* bias,
                   const float* res, float* oF, int M, int N, int K) {
    dim3 g(N / BN, M / BM);
    gemm_wmma_kernel<2><<<g, 256, 0, stream>>>(A, W, bias, res, oF, nullptr, M,
                                               N, K);
  };

  // --- self attention ---
  gemmH(tgt_h, wh[0], sa_bq, q1, BT, D, D, 0);
  gemmH(tgt_h, wh[1], sa_bk, k1, BT, D, D, 0);
  gemmH(tgt_h, wh[2], sa_bv, v1, BT, D, D, 0);
  attn_kernel<<<dim3(T / 128, B * HN), 256, 0, stream>>>(
      q1, k1, v1, at1, nullptr, nullptr, nullptr, tkpm, T, T, 0, scale);
  gemmF(at1, wh[3], sa_bo, tgt, ytmp, BT, D, D);
  ln_kernel<<<BT / 8, 256, 0, stream>>>(ytmp, ln1g, ln1b, x1, x1h, BT);

  // --- cross attention with analytic scale bias ---
  gemmH(x1h,   wh[4], ca_bq, q2, BT, D, D, 0);
  gemmH(mem_h, wh[5], ca_bk, k2, BS, D, D, 0);
  gemmH(mem_h, wh[6], ca_bv, v2, BS, D, D, 0);
  attn_kernel<<<dim3(T / 128, B * HN), 256, 0, stream>>>(
      q2, k2, v2, at2, qslog, quv, ksb, mkpm, T, S, 1, scale);
  gemmF(at2, wh[7], ca_bo, x1, ytmp, BT, D, D);
  ln_kernel<<<BT / 8, 256, 0, stream>>>(ytmp, ln2g, ln2b, x2, x2h, BT);

  // --- FFN ---
  gemmH(x2h, wh[8], b1, h1h, BT, F, D, 1);   // relu
  gemmF(h1h, wh[9], b2, x2, ytmp, BT, D, F);
  ln_kernel<<<BT / 8, 256, 0, stream>>>(ytmp, ln3g, ln3b, (float*)d_out, BT > 0 ? nullptr : nullptr, BT);
}
